// RecurrentGenerator_4209067950614
// MI455X (gfx1250) — matrix operand order repaired and checked by reading, not compiled, not run on hardware
//
#include <hip/hip_runtime.h>
#include <hip/hip_bf16.h>
#include <math.h>

// ---------------------------------------------------------------------------
// CDNA5 (gfx1250) stacked BiLSTM.
//  - Input projections: bf16 WMMA GEMM (v_wmma_f32_16x16x32_bf16), fp32 accum,
//    W tiles staged to LDS by the Tensor Data Mover (tensor_load_to_lds with
//    pad_enable -> 80B padded rows), double buffered, s_wait_tensorcnt sync.
//  - Recurrent scans: persistent blocks, batch split 2x16 per direction,
//    h state in LDS (bf16, padded rows), c state in registers (C-fragment
//    layout), W_hh streamed from L2, all 4 gates owned by the producing wave.
//  - Streaming buffers (xp, hbuf: >> 192MB L2) use non-temporal hints so L2
//    stays reserved for weights; activations (actb, 67MB, high reuse) stay RT.
// ---------------------------------------------------------------------------

typedef __bf16 bf16;
typedef __attribute__((ext_vector_type(16))) __bf16       v16bf;
typedef __attribute__((ext_vector_type(8)))  float        v8f;
typedef __attribute__((ext_vector_type(4)))  unsigned int u32x4;
typedef __attribute__((ext_vector_type(4)))  unsigned int v4u;
typedef __attribute__((ext_vector_type(4)))  int          v4i;
typedef __attribute__((ext_vector_type(8)))  int          v8i;

#define B_TOT 32
#define T_LEN 1024
#define H1    512
#define HP1   (H1 + 8)     // padded LDS row (bytes multiple of 16)
#define H2    64
#define HP2   (H2 + 8)
#define IN0   64
#define WROW  40           // staged W row stride in bf16 (80 bytes)

#if defined(__has_builtin)
#  if __has_builtin(__builtin_amdgcn_tensor_load_to_lds)
#    define HAVE_TDM 1
#  endif
#endif
#ifndef HAVE_TDM
#  define HAVE_TDM 0
#endif

union FragU { u32x4 q[2]; v16bf v; };

// A-fragment: elements 0..7 at p[0..7], elements 8..15 at p[16..23]
static __device__ __forceinline__ v16bf ld_frag_split(const bf16* p) {
  FragU f;
  f.q[0] = *(const u32x4*)(p);
  f.q[1] = *(const u32x4*)(p + 16);
  return f.v;
}
// B-fragment: elements 0..7 at p[0..7], elements 8..15 at p[16..23]
static __device__ __forceinline__ v16bf ld_frag_contig(const bf16* p) {
  FragU f;
  f.q[0] = *(const u32x4*)(p);
  f.q[1] = *(const u32x4*)(p + 16);
  return f.v;
}

static __device__ __forceinline__ v8f wmma_bf16(v16bf a, v16bf b, v8f c) {
  // (neg_a, A, neg_b, B, c_mod, C, reuse_a, reuse_b)
  return __builtin_amdgcn_wmma_f32_16x16x32_bf16(false, a, false, b,
                                                 (short)0, c, false, false);
}

// v_exp_f32 + v_rcp_f32 fast activations (no IEEE divide sequence on the
// serial recurrent path).
static __device__ __forceinline__ float sigm(float x) {
  return __builtin_amdgcn_rcpf(1.0f + __expf(-x));
}
static __device__ __forceinline__ float tanh_fast(float x) {
  // tanh(x) = 2*sigmoid(2x) - 1
  return 2.0f * __builtin_amdgcn_rcpf(1.0f + __expf(-2.0f * x)) - 1.0f;
}

#if HAVE_TDM
// Issue a TDM 2D tile load: 64 rows x 32 bf16 from row-major W (row stride =
// strideElems) into LDS at lds_byte_off, padded +16B after each 64B row so
// staged rows are 80B apart (bank-conflict-free ds_load_b128 fragments).
static __device__ __forceinline__ void tdm_stage_w(const bf16* gptr,
                                                   unsigned lds_byte_off,
                                                   int strideElems) {
  unsigned long long ga = (unsigned long long)(const void*)gptr;
  v4u g0;
  g0[0] = 1u;                                    // count=1 (valid descriptor)
  g0[1] = lds_byte_off;                          // lds_addr (bytes)
  g0[2] = (unsigned)(ga & 0xffffffffu);          // global_addr[31:0]
  g0[3] = (unsigned)((ga >> 32) & 0x01ffffffu)   // global_addr[56:32]
        | (2u << 30);                            // type = 2 ("image")
  v8i g1;
  g1[0] = (int)((1u << 16)        // data_size = 1 -> 2 bytes
              | (1u << 20)        // pad_enable
              | (3u << 22)        // pad_interval: every 16 dwords (one row)
              | (3u << 25));      // pad_amount: 4 dwords (16 bytes)
  g1[1] = (int)(((unsigned)strideElems & 0xffffu) << 16);   // tensor_dim0 lo
  g1[2] = (int)((((unsigned)strideElems >> 16) & 0xffffu)   // tensor_dim0 hi
              | (64u << 16));                               // tensor_dim1 = 64
  g1[3] = (int)(32u << 16);                                 // tile_dim0 = 32
  g1[4] = 64;                                               // tile_dim1 = 64
  g1[5] = strideElems;                                      // dim0_stride lo
  g1[6] = 0;
  g1[7] = 0;
  v4i z4 = (v4i)0;
#if __clang_major__ >= 23
  v8i z8 = (v8i)0;
  __builtin_amdgcn_tensor_load_to_lds(g0, g1, z4, z4, z8, 0);
#else
  __builtin_amdgcn_tensor_load_to_lds(g0, g1, z4, z4, 0);
#endif
}
#endif

// ---------------------------------------------------------------------------
__global__ __launch_bounds__(256)
void cast_f32_bf16(const float* __restrict__ s, bf16* __restrict__ d, long n) {
  long i = (long)blockIdx.x * blockDim.x + threadIdx.x;
  long stride = (long)gridDim.x * blockDim.x;
  for (; i < n; i += stride) d[i] = (bf16)__builtin_nontemporal_load(s + i);
}

// ---------------------------------------------------------------------------
// out[M,N] = A[M,K](bf16) @ W[N,K]^T(bf16) + b1[N] + b2[N]   (fp32 out, NT)
// block = 256 threads = 8 waves; wave computes 16 rows x 64 cols; the 64x32
// W tile (shared by all 8 waves) is staged once per k-slice via the TDM.
template <int K>
__global__ __launch_bounds__(256)
void gemm_bf16_wmma(const bf16* __restrict__ A, const bf16* __restrict__ W,
                    const float* __restrict__ b1, const float* __restrict__ b2,
                    float* __restrict__ out, int N) {
  __shared__ bf16 wstage[2][64 * WROW];          // 2 x 5120 B, 80 B row stride
  const int lane = threadIdx.x & 31;
  const int wave = threadIdx.x >> 5;
  const int halfsel = lane >> 4;
  const int nloc = lane & 15;
  const int row0 = blockIdx.y * 128 + wave * 16;
  const int col0 = blockIdx.x * 64;

  const bf16* Arow = A + (long)(row0 + nloc) * K;
  v8f acc[4] = {v8f{}, v8f{}, v8f{}, v8f{}};

  // ---- stage first W tile ----
#if HAVE_TDM
  if (threadIdx.x == 0) {
    tdm_stage_w(W + (long)col0 * K, (unsigned)(size_t)&wstage[0][0], K);
    __builtin_amdgcn_s_wait_tensorcnt(0);
  }
#else
  for (int idx = threadIdx.x; idx < 64 * 16; idx += 256) {
    int r = idx >> 4, c = idx & 15;
    ((unsigned*)&wstage[0][0])[r * (WROW / 2) + c] =
        ((const unsigned*)(W + (long)(col0 + r) * K))[c];
  }
#endif
  __syncthreads();

  int cur = 0;
#pragma unroll 2
  for (int k0 = 0; k0 < K; k0 += 32) {
    // stage next tile into the other buffer while computing this one
    if (k0 + 32 < K) {
#if HAVE_TDM
      if (threadIdx.x == 0)
        tdm_stage_w(W + (long)col0 * K + (k0 + 32),
                    (unsigned)(size_t)&wstage[cur ^ 1][0], K);
#else
      for (int idx = threadIdx.x; idx < 64 * 16; idx += 256) {
        int r = idx >> 4, c = idx & 15;
        ((unsigned*)&wstage[cur ^ 1][0])[r * (WROW / 2) + c] =
            ((const unsigned*)(W + (long)(col0 + r) * K + (k0 + 32)))[c];
      }
#endif
    }
    v16bf afrag = ld_frag_split(Arow + k0 + halfsel * 8);
    __builtin_prefetch(Arow + k0 + 64, 0, 0);     // speculative, unguarded
#pragma unroll
    for (int tc = 0; tc < 4; ++tc) {
      v16bf bfrag =
          ld_frag_contig(&wstage[cur][(tc * 16 + nloc) * WROW + halfsel * 8]);
      acc[tc] = wmma_bf16(afrag, bfrag, acc[tc]);
    }
#if HAVE_TDM
    if (threadIdx.x == 0) __builtin_amdgcn_s_wait_tensorcnt(0);
#endif
    __syncthreads();
    cur ^= 1;
  }

#pragma unroll
  for (int tc = 0; tc < 4; ++tc) {
    int n = col0 + tc * 16 + nloc;
    float bias = b1[n] + b2[n];
#pragma unroll
    for (int r = 0; r < 8; ++r) {
      int m = row0 + 8 * halfsel + r;       // C layout: M = r + 8*(lane>=16)
      __builtin_nontemporal_store(acc[tc][r] + bias, &out[(long)m * N + n]);
    }
  }
}

// ---------------------------------------------------------------------------
// Persistent recurrent scan, H=512 bidirectional layer.
// grid = 4 blocks: blockIdx.x = {dir, batch-half}; block = 1024 thr = 32 waves.
// Wave w owns hidden slice [16w,16w+16) for ALL FOUR gates (no gate exchange).
// k-loop kept rolled (unroll 1): 1 A-frag + 4 B-frags + 4 WMMAs live at a
// time, so nothing spills to scratch at 8 waves/SIMD.
__global__ __launch_bounds__(1024)
void lstm_scan512(const float* __restrict__ xp,   // [2][B,T,4H]
                  const bf16* __restrict__ Whh,   // [2][4H,H]
                  float* __restrict__ outbuf,     // [B,T,2H]
                  int T) {
  const int dir = blockIdx.x & 1;
  const int bh  = blockIdx.x >> 1;
  const bool rev = (dir == 1);
  const float* xpd = xp + (long)dir * B_TOT * T * 4 * H1;
  const bf16* Wd   = Whh + (long)dir * 4 * H1 * H1;

  const int lane = threadIdx.x & 31;
  const int wave = threadIdx.x >> 5;
  const int halfsel = lane >> 4;
  const int nloc = lane & 15;
  const int j0 = wave * 16;
  const int rA = lane & 15;               // A-fragment row = local batch

  __shared__ bf16 hsh[16 * HP1];          // bf16 h state, padded rows
  for (int i = threadIdx.x; i < 16 * HP1; i += blockDim.x)
    hsh[i] = (bf16)0.0f;
  v8f cst = {};                           // c state in C-fragment layout
  __syncthreads();

#pragma unroll 1
  for (int t = 0; t < T; ++t) {
    const int te = rev ? (T - 1 - t) : t;
    v8f acc[4];
#pragma unroll
    for (int g = 0; g < 4; ++g) {
      v8f a;
#pragma unroll
      for (int r = 0; r < 8; ++r) {
        int b = bh * 16 + 8 * halfsel + r;
        a[r] = __builtin_nontemporal_load(
            xpd + ((long)b * T + te) * (4 * H1) + g * H1 + j0 + nloc);
      }
      acc[g] = a;
    }
    // gates += h @ W_hh^T  (K = 512, 16 WMMA k-slices, 4 gate tiles each)
#pragma unroll 1
    for (int k0 = 0; k0 < H1; k0 += 32) {
      v16bf afrag = ld_frag_split(&hsh[rA * HP1 + k0 + halfsel * 8]);
#pragma unroll
      for (int g = 0; g < 4; ++g) {
        int n = g * H1 + j0 + nloc;
        const bf16* Wp = Wd + (long)n * H1 + k0 + halfsel * 8;
        v16bf bfrag = ld_frag_contig(Wp);
        acc[g] = wmma_bf16(afrag, bfrag, acc[g]);
      }
    }
    __syncthreads();                      // all LDS h reads done
#pragma unroll
    for (int r = 0; r < 8; ++r) {
      float iv = sigm(acc[0][r]);
      float fv = sigm(acc[1][r]);
      float gv = tanh_fast(acc[2][r]);
      float ov = sigm(acc[3][r]);
      float cv = fv * cst[r] + iv * gv;
      cst[r] = cv;
      float hv = ov * tanh_fast(cv);
      int bl = 8 * halfsel + r;
      int b = bh * 16 + bl;
      __builtin_nontemporal_store(
          hv, &outbuf[((long)b * T + te) * (2 * H1) + dir * H1 + j0 + nloc]);
      hsh[bl * HP1 + j0 + nloc] = (bf16)hv;
    }
    __syncthreads();                      // h visible for next step
  }
}

// ---------------------------------------------------------------------------
// Final unidirectional scan, H=64. grid = 2 batch-halves, 128 thr = 4 waves.
__global__ __launch_bounds__(128)
void lstm_scan64(const float* __restrict__ xp,    // [B,T,256]
                 const bf16* __restrict__ Whh,    // [256,64]
                 float* __restrict__ out,         // [B,T,64]  (x40)
                 int T) {
  const int bh = blockIdx.x;
  const int lane = threadIdx.x & 31;
  const int wave = threadIdx.x >> 5;
  const int halfsel = lane >> 4;
  const int nloc = lane & 15;
  const int j0 = wave * 16;
  const int rA = lane & 15;

  __shared__ bf16 hsh[16 * HP2];
  for (int i = threadIdx.x; i < 16 * HP2; i += blockDim.x)
    hsh[i] = (bf16)0.0f;
  v8f cst = {};
  __syncthreads();

#pragma unroll 1
  for (int t = 0; t < T; ++t) {
    v8f acc[4];
#pragma unroll
    for (int g = 0; g < 4; ++g) {
      v8f a;
#pragma unroll
      for (int r = 0; r < 8; ++r) {
        int b = bh * 16 + 8 * halfsel + r;
        a[r] = __builtin_nontemporal_load(
            xp + ((long)b * T + t) * 256 + g * H2 + j0 + nloc);
      }
      acc[g] = a;
    }
#pragma unroll 1
    for (int k0 = 0; k0 < H2; k0 += 32) {
      v16bf afrag = ld_frag_split(&hsh[rA * HP2 + k0 + halfsel * 8]);
#pragma unroll
      for (int g = 0; g < 4; ++g) {
        int n = g * H2 + j0 + nloc;
        v16bf bfrag = ld_frag_contig(Whh + (long)n * H2 + k0 + halfsel * 8);
        acc[g] = wmma_bf16(afrag, bfrag, acc[g]);
      }
    }
    __syncthreads();
#pragma unroll
    for (int r = 0; r < 8; ++r) {
      float iv = sigm(acc[0][r]);
      float fv = sigm(acc[1][r]);
      float gv = tanh_fast(acc[2][r]);
      float ov = sigm(acc[3][r]);
      float cv = fv * cst[r] + iv * gv;
      cst[r] = cv;
      float hv = ov * tanh_fast(cv);
      int bl = 8 * halfsel + r;
      int b = bh * 16 + bl;
      out[((long)b * T + t) * H2 + j0 + nloc] = 40.0f * hv;
      hsh[bl * HP2 + j0 + nloc] = (bf16)hv;
    }
    __syncthreads();
  }
}

// ---------------------------------------------------------------------------
extern "C" void kernel_launch(void* const* d_in, const int* in_sizes, int n_in,
                              void* d_out, int out_size, void* d_ws, size_t ws_size,
                              hipStream_t stream) {
  (void)in_sizes; (void)n_in; (void)out_size; (void)ws_size;
  const float* x     = (const float*)d_in[0];
  const float* Wih0  = (const float*)d_in[1];   // [2,2048,64]
  const float* Whh0  = (const float*)d_in[2];   // [2,2048,512]
  const float* bih0  = (const float*)d_in[3];   // [2,2048]
  const float* bhh0  = (const float*)d_in[4];
  const float* Wih12 = (const float*)d_in[5];   // [2,2,2048,1024]
  const float* Whh12 = (const float*)d_in[6];   // [2,2,2048,512]
  const float* bih12 = (const float*)d_in[7];   // [2,2,2048]
  const float* bhh12 = (const float*)d_in[8];
  const float* Wih2  = (const float*)d_in[9];   // [256,1024]
  const float* Whh2  = (const float*)d_in[10];  // [256,64]
  const float* bih2  = (const float*)d_in[11];
  const float* bhh2  = (const float*)d_in[12];
  float* out = (float*)d_out;

  const long BT = (long)B_TOT * T_LEN;          // 32768

  char* p = (char*)d_ws;
  auto take = [&](size_t bytes) -> void* {
    void* r = (void*)p;
    p += (bytes + 255) & ~(size_t)255;
    return r;
  };
  bf16*  xb     = (bf16*) take(BT * IN0 * 2);
  bf16*  actb   = (bf16*) take(BT * 1024 * 2);
  float* hbuf   = (float*)take(BT * 1024 * 4);
  float* xp     = (float*)take(2 * BT * 2048 * 4);
  float* xp2    = (float*)take(BT * 256 * 4);
  bf16*  Wih0b  = (bf16*) take(2L * 2048 * 64 * 2);
  bf16*  Whh0b  = (bf16*) take(2L * 2048 * 512 * 2);
  bf16*  Wih12b = (bf16*) take(4L * 2048 * 1024 * 2);
  bf16*  Whh12b = (bf16*) take(4L * 2048 * 512 * 2);
  bf16*  Wih2b  = (bf16*) take(256L * 1024 * 2);
  bf16*  Whh2b  = (bf16*) take(256L * 64 * 2);

  auto castN = [&](const float* s, bf16* d, long n) {
    long b = (n + 255) / 256;
    int blocks = (int)(b > 4096 ? 4096 : b);
    cast_f32_bf16<<<blocks, 256, 0, stream>>>(s, d, n);
  };

  // weight + input casts (deterministic each call)
  castN(x,     xb,     BT * IN0);
  castN(Wih0,  Wih0b,  2L * 2048 * 64);
  castN(Whh0,  Whh0b,  2L * 2048 * 512);
  castN(Wih12, Wih12b, 4L * 2048 * 1024);
  castN(Whh12, Whh12b, 4L * 2048 * 512);
  castN(Wih2,  Wih2b,  256L * 1024);
  castN(Whh2,  Whh2b,  256L * 64);

  dim3 blk(256);
  // ---- layer 0 (input 64) ----
  for (int d = 0; d < 2; ++d) {
    gemm_bf16_wmma<64><<<dim3(2048 / 64, (unsigned)(BT / 128)), blk, 0, stream>>>(
        xb, Wih0b + (long)d * 2048 * 64, bih0 + d * 2048, bhh0 + d * 2048,
        xp + (long)d * BT * 2048, 2048);
  }
  lstm_scan512<<<4, 1024, 0, stream>>>(xp, Whh0b, hbuf, T_LEN);
  castN(hbuf, actb, BT * 1024);

  // ---- layers 1,2 (input 1024) ----
  for (int l = 0; l < 2; ++l) {
    for (int d = 0; d < 2; ++d) {
      long wi = ((long)l * 2 + d);
      gemm_bf16_wmma<1024><<<dim3(2048 / 64, (unsigned)(BT / 128)), blk, 0, stream>>>(
          actb, Wih12b + wi * 2048 * 1024, bih12 + wi * 2048, bhh12 + wi * 2048,
          xp + (long)d * BT * 2048, 2048);
    }
    lstm_scan512<<<4, 1024, 0, stream>>>(
        xp, Whh12b + (long)l * 2 * 2048 * 512, hbuf, T_LEN);
    castN(hbuf, actb, BT * 1024);
  }

  // ---- rnn2 (input 1024, hidden 64, unidirectional) ----
  gemm_bf16_wmma<1024><<<dim3(256 / 64, (unsigned)(BT / 128)), blk, 0, stream>>>(
      actb, Wih2b, bih2, bhh2, xp2, 256);
  lstm_scan64<<<2, 128, 0, stream>>>(xp2, Whh2b, out, T_LEN);
}
